// RecurrentModel_87385404604454
// MI455X (gfx1250) — matrix operand order repaired and checked by reading, compiled, not run on hardware
//
#include <hip/hip_runtime.h>
#include <hip/hip_bf16.h>

// ---------------------------------------------------------------------------
// CDNA5 (gfx1250) bf16 WMMA types
// ---------------------------------------------------------------------------
typedef __attribute__((ext_vector_type(16))) __bf16 bf16x16;
typedef __attribute__((ext_vector_type(8)))  float  f32x8;

__device__ __forceinline__ float sigmoidf_(float x) {
    return 1.0f / (1.0f + __expf(-x));
}

// Issue 64B (4 x b128) of async global->LDS copy. Per ISA 10.x / 15.18.3 the
// instruction's IOFFSET is added to BOTH the LDS destination and the global
// source address, so one VGPR pair + offsets covers the whole 64B.
__device__ __forceinline__ void async_copy64(unsigned lds_dst, const void* gsrc) {
    asm volatile(
        "global_load_async_to_lds_b128 %0, %1, off\n\t"
        "global_load_async_to_lds_b128 %0, %1, off offset:16\n\t"
        "global_load_async_to_lds_b128 %0, %1, off offset:32\n\t"
        "global_load_async_to_lds_b128 %0, %1, off offset:48"
        :
        : "v"(lds_dst), "v"(gsrc)
        : "memory");
}

__device__ __forceinline__ void wait_asynccnt0() {
    asm volatile("s_wait_asynccnt 0x0" ::: "memory");
}

// ---------------------------------------------------------------------------
// Weight pre-conversion: f32 -> bf16 (tiny, once per call)
// ---------------------------------------------------------------------------
__global__ __launch_bounds__(256)
void w_to_bf16_kernel(const float* __restrict__ wf, __bf16* __restrict__ wb, int n) {
    const int i = blockIdx.x * 256 + threadIdx.x;
    if (i < n) wb[i] = (__bf16)wf[i];
}

// ---------------------------------------------------------------------------
// Input projection GEMM:  out[M,128] = in[M,K] @ W[128,K]^T + (bih + bhh)
// bf16 WMMA 16x16x32, f32 accumulate. 128 threads = 4 waves; each wave owns
// one 16-row M tile and all 8 N tiles. bf16 W k-chunks are double-buffered in
// LDS via async global->LDS (ASYNCcnt), overlapping staging with the WMMAs.
// ---------------------------------------------------------------------------
__global__ __launch_bounds__(128)
void xp_gemm_kernel(const float* __restrict__ in, int K,
                    const __bf16* __restrict__ Wb,    // [128, K] bf16
                    const float* __restrict__ bih,    // [128]
                    const float* __restrict__ bhh,    // [128]
                    float* __restrict__ out)          // [M, 128]
{
    __shared__ __bf16 ldsW[2][128][32];               // 2 x 8 KB double buffer

    const int tid  = threadIdx.x;
    const int lane = tid & 31;
    const int wave = tid >> 5;
    const int half = lane >> 4;
    const int l16  = lane & 15;
    const long mbase = ((long)blockIdx.x * 4 + wave) * 16;

    const unsigned ldsBase =
        (unsigned)(unsigned long long)(&ldsW[0][0][0]) + (unsigned)tid * 64u;
    const __bf16* wrow = Wb + (long)tid * K;          // this thread's W row

    f32x8 acc[8];
#pragma unroll
    for (int nt = 0; nt < 8; ++nt)
#pragma unroll
        for (int r = 0; r < 8; ++r) acc[nt][r] = 0.0f;

    const float* arowB = in + (mbase + l16) * (long)K;

    // prologue: stage chunk 0 into buffer 0
    async_copy64(ldsBase, wrow);
    wait_asynccnt0();
    __syncthreads();

    int buf = 0;
    for (int k0 = 0; k0 < K; k0 += 32) {
        // ---- kick off async staging of the next W chunk (other buffer) ---
        if (k0 + 32 < K)
            async_copy64(ldsBase + (unsigned)(buf ^ 1) * 8192u, wrow + k0 + 32);

        // ---- A fragment: row m = mbase+l16, K pattern per ISA 7.12.2 -----
        const float* arow = arowB + k0;
        const int kb = half * 8;
        if (k0 + 32 < K) __builtin_prefetch(arow + 32 + kb, 0, 1);
        float4 p0 = *(const float4*)(arow + kb);
        float4 p1 = *(const float4*)(arow + kb + 4);
        float4 p2 = *(const float4*)(arow + kb + 16);
        float4 p3 = *(const float4*)(arow + kb + 20);
        bf16x16 a;
        a[0]  = (__bf16)p0.x; a[1]  = (__bf16)p0.y; a[2]  = (__bf16)p0.z; a[3]  = (__bf16)p0.w;
        a[4]  = (__bf16)p1.x; a[5]  = (__bf16)p1.y; a[6]  = (__bf16)p1.z; a[7]  = (__bf16)p1.w;
        a[8]  = (__bf16)p2.x; a[9]  = (__bf16)p2.y; a[10] = (__bf16)p2.z; a[11] = (__bf16)p2.w;
        a[12] = (__bf16)p3.x; a[13] = (__bf16)p3.y; a[14] = (__bf16)p3.z; a[15] = (__bf16)p3.w;

        // ---- load all 8 B fragments, then chain the 8 WMMAs --------------
        const int kbb = half * 8;
        bf16x16 bfr[8];
#pragma unroll
        for (int nt = 0; nt < 8; ++nt) {
            const int n = nt * 16 + l16;
#pragma unroll
            for (int j = 0; j < 16; ++j) bfr[nt][j] = ldsW[buf][n][kbb + (j & 7) + ((j >> 3) << 4)];
        }
#pragma unroll
        for (int nt = 0; nt < 8; ++nt)
            acc[nt] = __builtin_amdgcn_wmma_f32_16x16x32_bf16(
                false, a, false, bfr[nt], (short)0, acc[nt], false, false);

        // ---- retire this wave's async copies, then swap buffers ----------
        wait_asynccnt0();
        __syncthreads();
        buf ^= 1;
    }

    // ---- epilogue: bias + store ------------------------------------------
#pragma unroll
    for (int nt = 0; nt < 8; ++nt) {
        const int n = nt * 16 + l16;
        const float bias = bih[n] + bhh[n];
#pragma unroll
        for (int r = 0; r < 8; ++r) {
            const int m = half * 8 + r;
            out[(mbase + m) * 128 + n] = acc[nt][r] + bias;
        }
    }
}

// ---------------------------------------------------------------------------
// Persistent LSTM recurrence: grid = (4 batch groups of 16, 2 directions),
// 256 threads = 8 waves, each wave owns one 16x16 N tile of the 128 gates.
// h@Whh^T is a single v_wmma_f32_16x16x32_bf16 per tile per timestep; the
// Whh B-fragment, per-row lengths and row base pointers are hoisted out of
// the 512-step time loop. Packed-sequence semantics via indexing.
// ---------------------------------------------------------------------------
__global__ __launch_bounds__(256)
void lstm_rec_kernel(const float* __restrict__ xpf,  // [B,T,128]
                     const float* __restrict__ xpb,  // [B,T,128]
                     const float* __restrict__ Whhf, // [128,32]
                     const float* __restrict__ Whhb, // [128,32]
                     const int*   __restrict__ lengths,
                     float* __restrict__ y,          // [B,T,64], pre-zeroed
                     int T)
{
    const int bg  = blockIdx.x;          // batch group: rows bg*16 .. bg*16+15
    const int dir = blockIdx.y;          // 0 = fwd, 1 = bwd
    const float* __restrict__ xp  = dir ? xpb  : xpf;
    const float* __restrict__ Whh = dir ? Whhb : Whhf;

    __shared__ float hS[16][32];
    __shared__ float cS[16][32];
    __shared__ float gS[16][128];

    const int tid  = threadIdx.x;
    const int lane = tid & 31;
    const int wave = tid >> 5;           // = N tile index 0..7
    const int half = lane >> 4;
    const int l16  = lane & 15;
    const int n    = wave * 16 + l16;    // gate column this lane produces

    // zero initial state
    for (int e = tid; e < 16 * 32; e += 256) {
        hS[e >> 5][e & 31] = 0.0f;
        cS[e >> 5][e & 31] = 0.0f;
    }

    // Whh B fragment (32x16 tile, column n), loaded once
    bf16x16 bfrag;
    {
        const int kbb = half * 16;
        const float* wr = Whh + n * 32 + kbb;
        float4 w0 = *(const float4*)(wr + 0);
        float4 w1 = *(const float4*)(wr + 4);
        float4 w2 = *(const float4*)(wr + 8);
        float4 w3 = *(const float4*)(wr + 12);
        bfrag[0]  = (__bf16)w0.x; bfrag[1]  = (__bf16)w0.y; bfrag[2]  = (__bf16)w0.z; bfrag[3]  = (__bf16)w0.w;
        bfrag[4]  = (__bf16)w1.x; bfrag[5]  = (__bf16)w1.y; bfrag[6]  = (__bf16)w1.z; bfrag[7]  = (__bf16)w1.w;
        bfrag[8]  = (__bf16)w2.x; bfrag[9]  = (__bf16)w2.y; bfrag[10] = (__bf16)w2.z; bfrag[11] = (__bf16)w2.w;
        bfrag[12] = (__bf16)w3.x; bfrag[13] = (__bf16)w3.y; bfrag[14] = (__bf16)w3.z; bfrag[15] = (__bf16)w3.w;
    }

    // loop-invariant per-row state for the C-fragment rows (m = half*8 + r)
    int lnr[8];
    const float* xprow[8];
#pragma unroll
    for (int r = 0; r < 8; ++r) {
        const int m = half * 8 + r;
        const int b = bg * 16 + m;
        lnr[r] = lengths[b];
        xprow[r] = xp + ((long)b * T) * 128 + n;
    }

    // loop-invariant state for this thread's two (m,hu) gate-update slots
    const int m0  = tid >> 5;            // 0..7
    const int m1  = m0 + 8;              // 8..15
    const int hu  = tid & 31;
    const int ln0 = lengths[bg * 16 + m0];
    const int ln1 = lengths[bg * 16 + m1];
    float* yrow0 = y + ((long)(bg * 16 + m0) * T) * 64 + dir * 32 + hu;
    float* yrow1 = y + ((long)(bg * 16 + m1) * T) * 64 + dir * 32 + hu;

    __syncthreads();

    for (int s = 0; s < T; ++s) {
        // ---- A fragment from h state (16x32, row m = l16) ----------------
        bf16x16 a;
        {
            const int kb = half * 16;
#pragma unroll
            for (int j = 0; j < 8; ++j) a[j]     = (__bf16)hS[l16][kb + j];
#pragma unroll
            for (int j = 0; j < 8; ++j) a[8 + j] = (__bf16)hS[l16][kb + 8 + j];
        }

        // ---- C initialized with precomputed x-projection (incl. biases) --
        f32x8 cacc;
#pragma unroll
        for (int r = 0; r < 8; ++r) {
            const int tt = dir ? ((s < lnr[r]) ? (lnr[r] - 1 - s) : 0) : s;
            cacc[r] = xprow[r][(long)tt * 128];
        }

        cacc = __builtin_amdgcn_wmma_f32_16x16x32_bf16(
            false, a, false, bfrag, (short)0, cacc, false, false);

#pragma unroll
        for (int r = 0; r < 8; ++r) gS[half * 8 + r][n] = cacc[r];
        __syncthreads();

        // ---- elementwise gate update (two fixed (m,hu) slots) ------------
        if (s < ln0) {
            const float c2 = sigmoidf_(gS[m0][32 + hu]) * cS[m0][hu] +
                             sigmoidf_(gS[m0][hu]) * tanhf(gS[m0][64 + hu]);
            const float h2 = sigmoidf_(gS[m0][96 + hu]) * tanhf(c2);
            cS[m0][hu] = c2;
            hS[m0][hu] = h2;
            const int pos = dir ? (ln0 - 1 - s) : s;
            yrow0[(long)pos * 64] = h2;
        }
        if (s < ln1) {
            const float c2 = sigmoidf_(gS[m1][32 + hu]) * cS[m1][hu] +
                             sigmoidf_(gS[m1][hu]) * tanhf(gS[m1][64 + hu]);
            const float h2 = sigmoidf_(gS[m1][96 + hu]) * tanhf(c2);
            cS[m1][hu] = c2;
            hS[m1][hu] = h2;
            const int pos = dir ? (ln1 - 1 - s) : s;
            yrow1[(long)pos * 64] = h2;
        }
        __syncthreads();
    }
}

// ---------------------------------------------------------------------------
// Mean pool over padded T + linear(2) + softmax. One block per batch row.
// ---------------------------------------------------------------------------
__global__ __launch_bounds__(256)
void pool_head_kernel(const float* __restrict__ y1,   // [B,T,64]
                      const float* __restrict__ Wlin, // [2,64]
                      const float* __restrict__ blin, // [2]
                      float* __restrict__ out,        // [B,2]
                      int T)
{
    const int b   = blockIdx.x;
    const int tid = threadIdx.x;
    const int hu  = tid & 63;
    const int tc  = tid >> 6;                 // 0..3

    float s = 0.0f;
    for (int t = tc; t < T; t += 4)
        s += y1[((long)b * T + t) * 64 + hu];

    __shared__ float part[4][64];
    __shared__ float pooled[64];
    part[tc][hu] = s;
    __syncthreads();
    if (tid < 64)
        pooled[tid] = (part[0][tid] + part[1][tid] +
                       part[2][tid] + part[3][tid]) / (float)T;
    __syncthreads();
    if (tid == 0) {
        float l0 = blin[0], l1 = blin[1];
        for (int k = 0; k < 64; ++k) {
            l0 += pooled[k] * Wlin[k];
            l1 += pooled[k] * Wlin[64 + k];
        }
        const float mx = fmaxf(l0, l1);
        const float e0 = __expf(l0 - mx), e1 = __expf(l1 - mx);
        const float inv = 1.0f / (e0 + e1);
        out[b * 2 + 0] = e0 * inv;
        out[b * 2 + 1] = e1 * inv;
    }
}

// ---------------------------------------------------------------------------
// Host-side orchestration (graph-capture safe: only kernel launches + async
// memsets on `stream`; workspace partitioned statically).
// ---------------------------------------------------------------------------
extern "C" void kernel_launch(void* const* d_in, const int* in_sizes, int n_in,
                              void* d_out, int out_size, void* d_ws, size_t ws_size,
                              hipStream_t stream) {
    (void)in_sizes; (void)n_in; (void)out_size; (void)ws_size;

    const float* x       = (const float*)d_in[0];
    const int*   lengths = (const int*)  d_in[1];
    const float* Wih0f = (const float*)d_in[2];
    const float* Whh0f = (const float*)d_in[3];
    const float* bih0f = (const float*)d_in[4];
    const float* bhh0f = (const float*)d_in[5];
    const float* Wih0b = (const float*)d_in[6];
    const float* Whh0b = (const float*)d_in[7];
    const float* bih0b = (const float*)d_in[8];
    const float* bhh0b = (const float*)d_in[9];
    const float* Wih1f = (const float*)d_in[10];
    const float* Whh1f = (const float*)d_in[11];
    const float* bih1f = (const float*)d_in[12];
    const float* bhh1f = (const float*)d_in[13];
    const float* Wih1b = (const float*)d_in[14];
    const float* Whh1b = (const float*)d_in[15];
    const float* bih1b = (const float*)d_in[16];
    const float* bhh1b = (const float*)d_in[17];
    const float* Wlin  = (const float*)d_in[18];
    const float* blin  = (const float*)d_in[19];
    float* out = (float*)d_out;

    const int  B = 64, T = 512, D = 768;
    const long M = (long)B * T;                       // 32768

    char*  ws  = (char*)d_ws;
    float* xpA = (float*)(ws);                        // [M,128] dir f (reused L1)
    float* xpB = (float*)(ws + M * 128 * 4);          // [M,128] dir b (reused L1)
    float* y0  = (float*)(ws + 2 * M * 128 * 4);      // [B,T,64]
    float* y1  = (float*)(ws + 2 * M * 128 * 4 + M * 64 * 4);
    __bf16* Wb0f = (__bf16*)(ws + 2 * M * 128 * 4 + 2 * M * 64 * 4);
    __bf16* Wb0b = Wb0f + 128 * 768;
    __bf16* Wb1f = Wb0b + 128 * 768;
    __bf16* Wb1b = Wb1f + 128 * 64;

    hipMemsetAsync(y0, 0, (size_t)(M * 64 * 4), stream);
    hipMemsetAsync(y1, 0, (size_t)(M * 64 * 4), stream);

    // weight pre-conversion to bf16 (tiny)
    w_to_bf16_kernel<<<(128 * 768 + 255) / 256, 256, 0, stream>>>(Wih0f, Wb0f, 128 * 768);
    w_to_bf16_kernel<<<(128 * 768 + 255) / 256, 256, 0, stream>>>(Wih0b, Wb0b, 128 * 768);
    w_to_bf16_kernel<<<(128 * 64 + 255) / 256, 256, 0, stream>>>(Wih1f, Wb1f, 128 * 64);
    w_to_bf16_kernel<<<(128 * 64 + 255) / 256, 256, 0, stream>>>(Wih1b, Wb1b, 128 * 64);

    const dim3 gemm_grid((unsigned)(M / 16 / 4));     // 512 blocks of 4 waves
    const dim3 gemm_blk(128);

    // layer 0
    xp_gemm_kernel<<<gemm_grid, gemm_blk, 0, stream>>>(x, D, Wb0f, bih0f, bhh0f, xpA);
    xp_gemm_kernel<<<gemm_grid, gemm_blk, 0, stream>>>(x, D, Wb0b, bih0b, bhh0b, xpB);
    lstm_rec_kernel<<<dim3(4, 2), 256, 0, stream>>>(xpA, xpB, Whh0f, Whh0b, lengths, y0, T);

    // layer 1 (input = y0, K = 64)
    xp_gemm_kernel<<<gemm_grid, gemm_blk, 0, stream>>>(y0, 64, Wb1f, bih1f, bhh1f, xpA);
    xp_gemm_kernel<<<gemm_grid, gemm_blk, 0, stream>>>(y0, 64, Wb1b, bih1b, bhh1b, xpB);
    lstm_rec_kernel<<<dim3(4, 2), 256, 0, stream>>>(xpA, xpB, Whh1f, Whh1b, lengths, y1, T);

    // head
    pool_head_kernel<<<64, 256, 0, stream>>>(y1, Wlin, blin, out, T);
}
